// SimpleDH_SFNN_86474871538266
// MI455X (gfx1250) — matrix operand order repaired and checked by reading, not compiled, not run on hardware
//
#include <hip/hip_runtime.h>
#include <hip/hip_bf16.h>

// ---------- problem constants (match reference) ----------
constexpr int kB = 128, kT = 256, kIN = 512, kH = 2048, kOUT = 128, kNB = 4;

// ---------- tiling ----------
constexpr int TILE_H = 32;            // h columns per workgroup (per branch)
constexpr int TILE_B = 64;            // batch rows per workgroup
constexpr int NCOL   = kNB * TILE_H;  // 128 GEMM columns in tile (4 branches x 32 h)
constexpr int LDK    = kIN + 8;       // padded bf16 row stride (16B pad -> bank spread)

typedef __attribute__((ext_vector_type(16))) __bf16         v16bf;
typedef __attribute__((ext_vector_type(8)))  float          v8f;
typedef __attribute__((ext_vector_type(4)))  float          f32x4;
typedef __attribute__((ext_vector_type(4)))  unsigned int   u32x4;
typedef __attribute__((ext_vector_type(4)))  unsigned short u16x4;

__device__ __forceinline__ unsigned short f2bf(float f) {
  unsigned int u = __float_as_uint(f);
  u += 0x7FFFu + ((u >> 16) & 1u);          // round-to-nearest-even
  return (unsigned short)(u >> 16);
}

// LDS byte offset of a __shared__ pointer: flat LDS aperture addresses carry the
// wave-relative LDS offset in their low 32 bits (ISA 10.2 aperture mapping).
__device__ __forceinline__ unsigned int lds_off(const void* p) {
  return (unsigned int)(size_t)p;
}

// CDNA5 async global->LDS copy, 16B per lane, tracked by ASYNCcnt.
__device__ __forceinline__ void async_ld_b128(unsigned int ldsaddr, const void* gaddr) {
  asm volatile("global_load_async_to_lds_b128 %0, %1, off"
               :: "v"(ldsaddr), "v"(gaddr) : "memory");
}
__device__ __forceinline__ void wait_async0() {
  asm volatile("s_wait_asynccnt 0x0" ::: "memory");
}

// ---------------- fp32 -> bf16 pre-pass (vectorized x4) ----------------
__global__ __launch_bounds__(256) void cvt_f32_bf16(const float* __restrict__ in,
                                                    unsigned short* __restrict__ out,
                                                    int n4) {
  const int i = blockIdx.x * 256 + threadIdx.x;
  if (i >= n4) return;
  const f32x4 f = ((const f32x4*)in)[i];
  u16x4 o;
  o.x = f2bf(f.x); o.y = f2bf(f.y); o.z = f2bf(f.z); o.w = f2bf(f.w);
  ((u16x4*)out)[i] = o;
}

// ---------------- fused GEMM + branch-leaky + LIF recurrence ----------------
// grid = (kH/TILE_H) * (kB/TILE_B) = 64 * 2 = 128 blocks, 256 threads (8 waves)
__global__ __launch_bounds__(256) void snn_fused(
    const unsigned short* __restrict__ xbf,   // [B][T][IN] bf16
    const unsigned short* __restrict__ wbf,   // [NB*H][IN] bf16
    const float* __restrict__ bias,           // [NB*H]
    const float* __restrict__ taus,           // [NB]
    float* __restrict__ spike_out)            // [B][H] fp32 (final-step spikes)
{
  __shared__ __attribute__((aligned(16))) unsigned short xlds[2][TILE_B][LDK];
  __shared__ __attribute__((aligned(16))) unsigned short wlds[NCOL][LDK];

  const int tid  = threadIdx.x;
  const int lane = tid & 31;
  const int wv   = tid >> 5;                      // wave 0..7
  const int wm   = wv & 3;                        // frag-row group (16 rows)
  const int wn   = wv >> 2;                       // h half (16 cols)
  const int g_h  = blockIdx.x & ((kH / TILE_H) - 1);
  const int g_b  = blockIdx.x / (kH / TILE_H);
  const int h0   = g_h * TILE_H;
  const int row0 = g_b * TILE_B;

  // --- async-stage W tile (resident all t) and x slice for t=0 ---
  for (int i = tid; i < NCOL * (kIN / 8); i += 256) {
    const int j  = i >> 6;                        // column 0..127 (= n*32 + hh)
    const int kc = i & 63;                        // 16B chunk along K
    const int n  = j >> 5;
    const int hh = j & (TILE_H - 1);
    async_ld_b128(lds_off(&wlds[j][kc * 8]),
                  wbf + ((size_t)(n * kH + h0 + hh)) * kIN + kc * 8);
  }
  for (int i = tid; i < TILE_B * (kIN / 8); i += 256) {
    const int r  = i >> 6;
    const int kc = i & 63;
    async_ld_b128(lds_off(&xlds[0][r][kc * 8]),
                  xbf + ((size_t)(row0 + r) * kT + 0) * kIN + kc * 8);
  }

  // per-lane column metadata (branch sum is lane-local by construction)
  const int hloc = wn * 16 + (lane & 15);         // h within stripe
  const int hcol = h0 + hloc;                     // global h
  float alpha[kNB], onema[kNB], bcol[kNB];
  #pragma unroll
  for (int n = 0; n < kNB; ++n) {
    const float a = 1.0f / (1.0f + __expf(-taus[n]));   // sigmoid(tau)
    alpha[n] = a; onema[n] = 1.0f - a;
    bcol[n]  = bias[n * kH + hcol];
  }

  // A/B fragment LDS addressing (ISA 16-bit A 16x32 / B 32x16 layouts)
  const int mloc = wm * 16 + (lane & 15);         // A-matrix row in tile
  const int kbA  = (lane >> 4) * 8;               // A: K[0..8)|K[8..16) then +16
  const int kbB  = (lane >> 4) * 8;               // B: K[0..8)|K[8..16) then +16

  v8f s[kNB], vmem;
  #pragma unroll
  for (int n = 0; n < kNB; ++n)
    #pragma unroll
    for (int q = 0; q < 8; ++q) s[n][q] = 0.0f;
  #pragma unroll
  for (int q = 0; q < 8; ++q) vmem[q] = 0.0f;

  wait_async0();
  __syncthreads();

  union AV { u32x4 u[2]; v16bf v; };

  for (int t = 0; t < kT; ++t) {
    const int cur = t & 1;

    // issue async staging of x[:, t+1, :] into the other buffer; no wait yet —
    // it overlaps the 64 WMMAs below and is waited at the end of this step.
    if (t + 1 < kT) {
      for (int i = tid; i < TILE_B * (kIN / 8); i += 256) {
        const int r  = i >> 6;
        const int kc = i & 63;
        async_ld_b128(lds_off(&xlds[cur ^ 1][r][kc * 8]),
                      xbf + ((size_t)(row0 + r) * kT + (t + 1)) * kIN + kc * 8);
      }
    }

    v8f acc[kNB];
    #pragma unroll
    for (int n = 0; n < kNB; ++n)
      #pragma unroll
      for (int q = 0; q < 8; ++q) acc[n][q] = 0.0f;

    // 2-deep software pipeline: fragments for kc+1 are in flight while the
    // four WMMAs of kc issue (breaks the dscnt==0 stall chain).
    AV afrag[2]; AV bfrag[2][kNB];
    {
      afrag[0].u[0] = *(const u32x4*)(&xlds[cur][mloc][kbA]);
      afrag[0].u[1] = *(const u32x4*)(&xlds[cur][mloc][kbA + 16]);
      #pragma unroll
      for (int n = 0; n < kNB; ++n) {
        const unsigned short* p = &wlds[n * TILE_H + hloc][kbB];
        bfrag[0][n].u[0] = *(const u32x4*)p;
        bfrag[0][n].u[1] = *(const u32x4*)(p + 16);
      }
    }
    #pragma unroll
    for (int kc = 0; kc < kIN / 32; ++kc) {
      const int cs = kc & 1;
      if (kc + 1 < kIN / 32) {
        const int ns = cs ^ 1;
        const int ko = (kc + 1) * 32;
        afrag[ns].u[0] = *(const u32x4*)(&xlds[cur][mloc][ko + kbA]);
        afrag[ns].u[1] = *(const u32x4*)(&xlds[cur][mloc][ko + kbA + 16]);
        #pragma unroll
        for (int n = 0; n < kNB; ++n) {
          const unsigned short* p = &wlds[n * TILE_H + hloc][ko + kbB];
          bfrag[ns][n].u[0] = *(const u32x4*)p;
          bfrag[ns][n].u[1] = *(const u32x4*)(p + 16);
        }
      }
      #pragma unroll
      for (int n = 0; n < kNB; ++n) {
        acc[n] = __builtin_amdgcn_wmma_f32_16x16x32_bf16(
            false, afrag[cs].v, false, bfrag[cs][n].v, (short)0, acc[n], false, false);
      }
    }

    // recurrence: all four branches of (b,h) live in the SAME lane/register slot
    #pragma unroll
    for (int q = 0; q < 8; ++q) {
      float comb = 0.0f;
      #pragma unroll
      for (int n = 0; n < kNB; ++n) {
        const float bi = acc[n][q] + bcol[n];               // branch linear out
        const float sn = alpha[n] * s[n][q] + onema[n] * bi;
        s[n][q] = sn;
        comb += sn;
      }
      float v = vmem[q];
      v = v + (comb - v) * 0.5f;                            // TAU_LIF = 2
      const float sp = (v >= 1.0f) ? 1.0f : 0.0f;           // heaviside(v - VTH)
      vmem[q] = v * (1.0f - sp);                            // hard reset
      if (t == kT - 1) {
        const int brow = row0 + wm * 16 + q + 8 * (lane >> 4);
        spike_out[(size_t)brow * kH + hcol] = sp;
      }
    }

    // staged t+1 complete + all waves done reading buffer `cur`
    wait_async0();
    __syncthreads();
  }
}

// ---------------- output head on final-step spikes (tiny, fp32) ----------------
__global__ __launch_bounds__(kOUT) void out_head(const float* __restrict__ spike,
                                                 const float* __restrict__ out_w,
                                                 const float* __restrict__ out_b,
                                                 float* __restrict__ out) {
  __shared__ float srow[kH];
  const int b = blockIdx.x, o = threadIdx.x;
  for (int i = o; i < kH; i += kOUT) srow[i] = spike[(size_t)b * kH + i];
  __syncthreads();
  float acc = out_b[o];
  for (int k = 0; k < kH; ++k) acc = fmaf(srow[k], out_w[(size_t)o * kH + k], acc);
  out[b * kOUT + o] = acc;
}

extern "C" void kernel_launch(void* const* d_in, const int* in_sizes, int n_in,
                              void* d_out, int out_size, void* d_ws, size_t ws_size,
                              hipStream_t stream) {
  (void)in_sizes; (void)n_in; (void)out_size; (void)ws_size;
  const float* x     = (const float*)d_in[0];   // [B,T,IN]
  const float* W     = (const float*)d_in[1];   // [NB,H,IN]
  const float* bias  = (const float*)d_in[2];   // [NB,H]
  const float* taus  = (const float*)d_in[3];   // [NB]
  const float* out_w = (const float*)d_in[4];   // [OUT,H]
  const float* out_b = (const float*)d_in[5];   // [OUT]
  float* out = (float*)d_out;                   // [B,OUT]

  // workspace: x_bf16 (32MB) | W_bf16 (8MB) | spike fp32 (1MB)  => ~42MB
  const size_t nx = (size_t)kB * kT * kIN;
  const size_t nw = (size_t)kNB * kH * kIN;
  unsigned short* xbf = (unsigned short*)d_ws;
  unsigned short* wbf = xbf + nx;
  float* spike = (float*)(wbf + nw);

  cvt_f32_bf16<<<(int)(nx / 4 / 256), 256, 0, stream>>>(x, xbf, (int)(nx / 4));
  cvt_f32_bf16<<<(int)(nw / 4 / 256), 256, 0, stream>>>(W, wbf, (int)(nw / 4));

  dim3 grid((kH / TILE_H) * (kB / TILE_B));     // 64 h-stripes x 2 batch halves = 128 WGs
  snn_fused<<<grid, 256, 0, stream>>>(xbf, wbf, bias, taus, spike);
  out_head<<<kB, kOUT, 0, stream>>>(spike, out_w, out_b, out);
}
